// NODESolver_61297773248706
// MI455X (gfx1250) — compile-verified, matrix operand order checked by reading, not run on hardware
//
#include <hip/hip_runtime.h>
#include <hip/hip_bf16.h>

typedef __attribute__((ext_vector_type(16))) __bf16 v16bf;
typedef __attribute__((ext_vector_type(8)))  __bf16 v8bf;
typedef __attribute__((ext_vector_type(8)))  float  v8f;

// ---------------------------------------------------------------------------
// fast tanh: v_exp_f32 + v_rcp_f32
// ---------------------------------------------------------------------------
__device__ __forceinline__ float ftanh(float x) {
    x = fminf(fmaxf(x, -15.0f), 15.0f);
    float e = __builtin_amdgcn_exp2f(x * 2.8853900817779268f); // 2*log2(e)
    float r = __builtin_amdgcn_rcpf(e + 1.0f);
    return (e - 1.0f) * r;
}

// ---------------------------------------------------------------------------
// One wave computes a 16xN tile (N = NSUBS*16) of  D = A(16xK) * W^T + bias.
// A: LDS, row-major bf16, padded stride (elements). W: pre-packed bf16 in the
// exact WMMA B-fragment layout: tiles of 512 elems, index
//   ((kt*ntilesN + ntile)<<9) + lane*16 + i
// C/D frag layout (ISA 7.12.2): acc[r] = D[M = r + 8*(lane>>4), N = lane&15].
//
// k-loop: explicit ping-pong double buffer, two k-tiles per iteration
// (kts is even at every call site), `#pragma unroll 1` so the step-invariant
// weight fragments can never be hoisted/spilled; no register rotation movs.
// ---------------------------------------------------------------------------
template <int NSUBS, bool ACT, bool TOLDS>
__device__ __forceinline__ void gemm_tile(
    const __bf16* __restrict__ A, int Astride, int Kdim,
    const __bf16* __restrict__ Wp, int ntilesN,
    const float* __restrict__ bias, int nbase0,
    __bf16* Hout, int Hstride,
    float kout[][8], int lane)
{
    const int ln = lane & 15;
    const int hi = lane >> 4;
    v8f acc[NSUBS];
    const v8f z8 = {0.f, 0.f, 0.f, 0.f, 0.f, 0.f, 0.f, 0.f};
#pragma unroll
    for (int ns = 0; ns < NSUBS; ++ns) acc[ns] = z8;

    const int kts = Kdim >> 5;               // 8 or 16 (even)
    const int ktstride = ntilesN << 9;       // elements between k-tiles
    const __bf16* wbase = Wp + (((size_t)nbase0 >> 4) << 9) + lane * 16;
    const __bf16* arow  = A + ln * Astride + hi * 8;

    auto loadA = [&](int kt) -> v16bf {
        const __bf16* ap = arow + kt * 32;
        v8bf lo = *(const v8bf*)(ap);
        v8bf hc = *(const v8bf*)(ap + 16);
        return __builtin_shufflevector(lo, hc, 0, 1, 2, 3, 4, 5, 6, 7,
                                       8, 9, 10, 11, 12, 13, 14, 15);
    };
    auto loadB = [&](int kt, v16bf (&b)[NSUBS]) {
        const __bf16* wk = wbase + (size_t)kt * ktstride;
#pragma unroll
        for (int ns = 0; ns < NSUBS; ++ns)
            b[ns] = *(const v16bf*)(wk + ns * 512);
    };
    auto mult = [&](v16bf a, v16bf (&b)[NSUBS]) {
#pragma unroll
        for (int ns = 0; ns < NSUBS; ++ns)
            acc[ns] = __builtin_amdgcn_wmma_f32_16x16x32_bf16(
                false, a, false, b[ns], (short)0, acc[ns], false, false);
    };

    v16bf a0, a1;
    v16bf b0[NSUBS], b1[NSUBS];

    // prologue: k-tile 0 -> buffer 0
    a0 = loadA(0);
    loadB(0, b0);

#pragma unroll 1
    for (int kt = 0; kt + 2 < kts; kt += 2) {
        a1 = loadA(kt + 1); loadB(kt + 1, b1);   // prefetch odd tile
        mult(a0, b0);                            // multiply even tile
        a0 = loadA(kt + 2); loadB(kt + 2, b0);   // prefetch next even tile
        mult(a1, b1);                            // multiply odd tile
    }
    // epilogue: buffers hold kts-2 (b0) and we still need kts-1
    a1 = loadA(kts - 1); loadB(kts - 1, b1);
    mult(a0, b0);
    mult(a1, b1);

#pragma unroll
    for (int ns = 0; ns < NSUBS; ++ns) {
        float bv = bias[nbase0 + ns * 16 + ln];
        union { v8f v; float f[8]; } u;
        u.v = acc[ns];
#pragma unroll
        for (int r = 0; r < 8; ++r) {
            float val = u.f[r] + bv;
            if constexpr (ACT) val = ftanh(val);
            if constexpr (TOLDS)
                Hout[(r + 8 * hi) * Hstride + nbase0 + ns * 16 + ln] = (__bf16)val;
            else
                kout[ns][r] = val;
        }
    }
}

// ---------------------------------------------------------------------------
// Weight pack: fp32 (N,K) row-major  ->  bf16 WMMA-B fragment layout.
// packed[p]: i=p&15, lane=(p>>4)&31, tile=p>>9, ntile=tile%ntiles,
// kt=tile/ntiles; value = W[ntile*16 + (lane&15)][kt*32 + Kf(i,lane>>4)]
// with Kf = (i&8)*2 + (lane>>4)*8 + (i&7)   (ISA 16-bit A/B layout)
// ---------------------------------------------------------------------------
__global__ __launch_bounds__(256) void pack_kernel(const float* __restrict__ src,
                                                   __bf16* __restrict__ dst,
                                                   int N, int K)
{
    int p = blockIdx.x * 256 + threadIdx.x;
    int i = p & 15;
    int lane = (p >> 4) & 31;
    int tile = p >> 9;
    int ntiles = N >> 4;
    int ntile = tile % ntiles;
    int kt = tile / ntiles;
    int hi = lane >> 4;
    int n = ntile * 16 + (lane & 15);
    int k = kt * 32 + ((i & 8) * 2) + hi * 8 + (i & 7);
    dst[p] = (__bf16)src[(size_t)n * K + k];
}

// ---------------------------------------------------------------------------
// Encoder: only columns 0..3 of y influence z0 = x[:,:,0].
// layer: x = tanh?( conv1x1(x)+rb + conv3(x)+cb )
// ---------------------------------------------------------------------------
__global__ __launch_bounds__(256) void enc0_kernel(
    const float* __restrict__ y, const float* __restrict__ ew0,
    const float* __restrict__ eb0, const float* __restrict__ rw0,
    const float* __restrict__ rb0, float* __restrict__ x1)
{
    int idx = blockIdx.x * 256 + threadIdx.x;   // B*256*3
    int l = idx % 3;
    int ch = (idx / 3) & 255;
    int b = idx / 768;
    float s = eb0[ch] + rb0[ch];
    for (int c = 0; c < 64; ++c) {
        const float* yb = y + ((size_t)b * 64 + c) * 1024;
        const float* w = ew0 + ((size_t)ch * 64 + c) * 3;
        float y0 = yb[l];
        s += rw0[ch * 64 + c] * y0;
        if (l > 0) s += w[0] * yb[l - 1];
        s += w[1] * y0;
        s += w[2] * yb[l + 1];
    }
    x1[((size_t)b * 256 + ch) * 3 + l] = ftanh(s);
}

__global__ __launch_bounds__(256) void enc1_kernel(
    const float* __restrict__ x1, const float* __restrict__ ew1,
    const float* __restrict__ eb1, const float* __restrict__ rw1,
    const float* __restrict__ rb1, float* __restrict__ x2)
{
    int idx = blockIdx.x * 256 + threadIdx.x;   // B*256*2
    int l = idx & 1;
    int ch = (idx >> 1) & 255;
    int b = idx >> 9;
    float s = eb1[ch] + rb1[ch];
    for (int c = 0; c < 256; ++c) {
        const float* xb = x1 + ((size_t)b * 256 + c) * 3;
        const float* w = ew1 + ((size_t)ch * 256 + c) * 3;
        float x0 = xb[l];
        s += rw1[ch * 256 + c] * x0;
        if (l > 0) s += w[0] * xb[l - 1];
        s += w[1] * x0;
        s += w[2] * xb[l + 1];
    }
    x2[((size_t)b * 256 + ch) * 2 + l] = ftanh(s);
}

__global__ __launch_bounds__(256) void enc2_kernel(
    const float* __restrict__ x2, const float* __restrict__ ew2,
    const float* __restrict__ eb2, const float* __restrict__ rw2,
    const float* __restrict__ rb2, float* __restrict__ z0g,
    __bf16* __restrict__ zs)
{
    int idx = blockIdx.x * 256 + threadIdx.x;   // B*256
    int ch = idx & 255;
    int b = idx >> 8;
    float s = eb2[ch] + rb2[ch];
    for (int c = 0; c < 256; ++c) {
        const float* xb = x2 + ((size_t)b * 256 + c) * 2;
        const float* w = ew2 + ((size_t)ch * 256 + c) * 3;
        s += rw2[ch * 256 + c] * xb[0];
        s += w[1] * xb[0] + w[2] * xb[1];   // w[0] hits left zero-pad
    }
    z0g[(size_t)b * 256 + ch] = s;          // last layer: no tanh
    zs[(size_t)b * 256 + ch] = (__bf16)s;   // pred_z[t=0]
}

// ---------------------------------------------------------------------------
// Persistent RK4 ODE kernel: 16 workgroups, one 16-row batch tile each.
// 8 waves: wave w owns N cols [64w,64w+64) in 512-wide layers, [32w,32w+32)
// in the 256-wide output layer. State & accumulators live in registers in
// WMMA C/D fragment layout; activations round-trip through padded LDS bf16.
// ---------------------------------------------------------------------------
__global__ __launch_bounds__(256) void ode_kernel(
    const float* __restrict__ z0g, const float* __restrict__ tarr,
    const __bf16* __restrict__ pfw0, const float* __restrict__ fb0,
    const __bf16* __restrict__ pfw1, const float* __restrict__ fb1,
    const __bf16* __restrict__ pfw2, const float* __restrict__ fb2,
    __bf16* __restrict__ zs)
{
    __shared__ __align__(16) __bf16 zin[16 * 264];   // 16 x 256 (+pad)
    __shared__ __align__(16) __bf16 h1[16 * 520];    // 16 x 512 (+pad)
    __shared__ __align__(16) __bf16 h2[16 * 520];

    const int tid = threadIdx.x;
    const int wave = tid >> 5;
    const int lane = tid & 31;
    const int ln = lane & 15;
    const int hi = lane >> 4;
    const int mbase = blockIdx.x * 16;

    float zb[2][8], accum[2][8], ztmp[2][8], kf[2][8];

    // load initial state into C/D-fragment registers
#pragma unroll
    for (int ns = 0; ns < 2; ++ns) {
        int col = wave * 32 + ns * 16 + ln;
#pragma unroll
        for (int r = 0; r < 8; ++r)
            zb[ns][r] = z0g[(size_t)(mbase + r + 8 * hi) * 256 + col];
    }

    auto write_state = [&](float s[2][8]) {
#pragma unroll
        for (int ns = 0; ns < 2; ++ns) {
            int col = wave * 32 + ns * 16 + ln;
#pragma unroll
            for (int r = 0; r < 8; ++r)
                zin[(r + 8 * hi) * 264 + col] = (__bf16)s[ns][r];
        }
    };

    auto rhs = [&](float out[2][8]) {
        __syncthreads();   // zin visible to all
        gemm_tile<4, true, true>(zin, 264, 256, pfw0, 32, fb0, wave * 64,
                                 h1, 520, (float(*)[8])nullptr, lane);
        __syncthreads();
        gemm_tile<4, true, true>(h1, 520, 512, pfw1, 32, fb1, wave * 64,
                                 h2, 520, (float(*)[8])nullptr, lane);
        __syncthreads();
        gemm_tile<2, false, false>(h2, 520, 512, pfw2, 16, fb2, wave * 32,
                                   (__bf16*)nullptr, 0, out, lane);
    };

    write_state(zb);

#pragma unroll 1
    for (int step = 0; step < 199; ++step) {
        float t0 = tarr[step], t1 = tarr[step + 1];
        float dt = t1 - t0;

        rhs(kf);                                         // k1
#pragma unroll
        for (int ns = 0; ns < 2; ++ns)
#pragma unroll
            for (int r = 0; r < 8; ++r) {
                accum[ns][r] = kf[ns][r];
                ztmp[ns][r] = zb[ns][r] + 0.5f * dt * kf[ns][r];
            }
        write_state(ztmp);

        rhs(kf);                                         // k2
#pragma unroll
        for (int ns = 0; ns < 2; ++ns)
#pragma unroll
            for (int r = 0; r < 8; ++r) {
                accum[ns][r] += 2.0f * kf[ns][r];
                ztmp[ns][r] = zb[ns][r] + 0.5f * dt * kf[ns][r];
            }
        write_state(ztmp);

        rhs(kf);                                         // k3
#pragma unroll
        for (int ns = 0; ns < 2; ++ns)
#pragma unroll
            for (int r = 0; r < 8; ++r) {
                accum[ns][r] += 2.0f * kf[ns][r];
                ztmp[ns][r] = zb[ns][r] + dt * kf[ns][r];
            }
        write_state(ztmp);

        rhs(kf);                                         // k4
        float c6 = dt * (1.0f / 6.0f);
#pragma unroll
        for (int ns = 0; ns < 2; ++ns)
#pragma unroll
            for (int r = 0; r < 8; ++r)
                zb[ns][r] += c6 * (accum[ns][r] + kf[ns][r]);
        write_state(zb);

        // save pred_z[step+1] for the decoder
#pragma unroll
        for (int ns = 0; ns < 2; ++ns) {
            int col = wave * 32 + ns * 16 + ln;
#pragma unroll
            for (int r = 0; r < 8; ++r)
                zs[(size_t)(step + 1) * 65536 +
                   (size_t)(mbase + r + 8 * hi) * 256 + col] = (__bf16)zb[ns][r];
        }
    }
}

// ---------------------------------------------------------------------------
// Decoder: M = 200*256 = 51200 rows, 256 ->512(tanh) ->512(tanh) ->64.
// 3200 workgroups x 16 rows, 8 waves x 64-col slices (NSUBS=4) for the
// 512-wide layers; waves 0..3 handle the 4 16-col subtiles of the final
// 64-wide layer. Output written directly transposed to (B, SIG, T).
// ---------------------------------------------------------------------------
__global__ __launch_bounds__(256) void dec_kernel(
    const __bf16* __restrict__ zs,
    const __bf16* __restrict__ pdw0, const float* __restrict__ db0,
    const __bf16* __restrict__ pdw1, const float* __restrict__ db1,
    const __bf16* __restrict__ pdw2, const float* __restrict__ db2,
    float* __restrict__ out)
{
    __shared__ __align__(16) __bf16 a0[16 * 264];
    __shared__ __align__(16) __bf16 h1[16 * 520];
    __shared__ __align__(16) __bf16 h2[16 * 520];

    const int tid = threadIdx.x;
    const int wave = tid >> 5;
    const int lane = tid & 31;
    const int ln = lane & 15;
    const int hi = lane >> 4;
    const int rbase = blockIdx.x * 16;

    // stage 16x256 bf16 input rows into LDS (8-elem vector copies)
    for (int c = tid; c < 16 * 32; c += 256) {
        int r = c >> 5;
        int cc = (c & 31) * 8;
        *(v8bf*)(a0 + r * 264 + cc) = *(const v8bf*)(zs + (size_t)(rbase + r) * 256 + cc);
    }
    __syncthreads();

    // d0: 256 -> 512, tanh
    gemm_tile<4, true, true>(a0, 264, 256, pdw0, 32, db0, wave * 64,
                             h1, 520, (float(*)[8])nullptr, lane);
    __syncthreads();

    // d1: 512 -> 512, tanh
    gemm_tile<4, true, true>(h1, 520, 512, pdw1, 32, db1, wave * 64,
                             h2, 520, (float(*)[8])nullptr, lane);
    __syncthreads();

    // d2: 512 -> 64, no act; waves 0..3 take one 16-col subtile each
    if (wave < 4) {
        float kf[1][8];
        gemm_tile<1, false, false>(h2, 520, 512, pdw2, 4, db2,
                                   wave * 16, (__bf16*)nullptr, 0, kf, lane);
        int s = wave * 16 + ln;
#pragma unroll
        for (int r = 0; r < 8; ++r) {
            int m = rbase + r + 8 * hi;   // m = t*256 + b
            int t = m >> 8;
            int b = m & 255;
            out[((size_t)b * 64 + s) * 200 + t] = kf[0][r];
        }
    }
}

// ---------------------------------------------------------------------------
extern "C" void kernel_launch(void* const* d_in, const int* in_sizes, int n_in,
                              void* d_out, int out_size, void* d_ws, size_t ws_size,
                              hipStream_t stream)
{
    const float* y    = (const float*)d_in[0];
    const float* tarr = (const float*)d_in[1];
    const float* ew0  = (const float*)d_in[2];
    const float* eb0  = (const float*)d_in[3];
    const float* rw0  = (const float*)d_in[4];
    const float* rb0  = (const float*)d_in[5];
    const float* ew1  = (const float*)d_in[6];
    const float* eb1  = (const float*)d_in[7];
    const float* rw1  = (const float*)d_in[8];
    const float* rb1  = (const float*)d_in[9];
    const float* ew2  = (const float*)d_in[10];
    const float* eb2  = (const float*)d_in[11];
    const float* rw2  = (const float*)d_in[12];
    const float* rb2  = (const float*)d_in[13];
    const float* fw0  = (const float*)d_in[14];
    const float* fb0  = (const float*)d_in[15];
    const float* fw1  = (const float*)d_in[16];
    const float* fb1  = (const float*)d_in[17];
    const float* fw2  = (const float*)d_in[18];
    const float* fb2  = (const float*)d_in[19];
    const float* dw0  = (const float*)d_in[20];
    const float* db0  = (const float*)d_in[21];
    const float* dw1  = (const float*)d_in[22];
    const float* db1  = (const float*)d_in[23];
    const float* dw2  = (const float*)d_in[24];
    const float* db2  = (const float*)d_in[25];
    float* out = (float*)d_out;

    char* ws = (char*)d_ws;
    size_t off = 0;
    auto walloc = [&](size_t bytes) -> void* {
        void* p = ws + off;
        off += (bytes + 255) & ~(size_t)255;
        return p;
    };

    float*  z0g  = (float*)walloc(65536 * 4);          // (B, LAT)
    float*  x1   = (float*)walloc(196608 * 4);         // (B, 256, 3)
    float*  x2   = (float*)walloc(131072 * 4);         // (B, 256, 2)
    __bf16* zs   = (__bf16*)walloc((size_t)200 * 65536 * 2); // (T, B, LAT) bf16
    __bf16* pfw0 = (__bf16*)walloc(131072 * 2);        // 512x256
    __bf16* pfw1 = (__bf16*)walloc(262144 * 2);        // 512x512
    __bf16* pfw2 = (__bf16*)walloc(131072 * 2);        // 256x512
    __bf16* pdw0 = (__bf16*)walloc(131072 * 2);        // 512x256
    __bf16* pdw1 = (__bf16*)walloc(262144 * 2);        // 512x512
    __bf16* pdw2 = (__bf16*)walloc(32768 * 2);         // 64x512

    // pack weights into WMMA-B fragment layout (bf16)
    pack_kernel<<<512,  256, 0, stream>>>(fw0, pfw0, 512, 256);
    pack_kernel<<<1024, 256, 0, stream>>>(fw1, pfw1, 512, 512);
    pack_kernel<<<512,  256, 0, stream>>>(fw2, pfw2, 256, 512);
    pack_kernel<<<512,  256, 0, stream>>>(dw0, pdw0, 512, 256);
    pack_kernel<<<1024, 256, 0, stream>>>(dw1, pdw1, 512, 512);
    pack_kernel<<<128,  256, 0, stream>>>(dw2, pdw2, 64, 512);

    // encoder (only cols 0..3 of y matter for z0)
    enc0_kernel<<<768, 256, 0, stream>>>(y, ew0, eb0, rw0, rb0, x1);
    enc1_kernel<<<512, 256, 0, stream>>>(x1, ew1, eb1, rw1, rb1, x2);
    enc2_kernel<<<256, 256, 0, stream>>>(x2, ew2, eb2, rw2, rb2, z0g, zs);

    // RK4 integration (persistent, 16 independent batch tiles)
    ode_kernel<<<16, 256, 0, stream>>>(z0g, tarr, pfw0, fb0, pfw1, fb1,
                                       pfw2, fb2, zs);

    // decoder + transposed store
    dec_kernel<<<3200, 256, 0, stream>>>(zs, pdw0, db0, pdw1, db1,
                                         pdw2, db2, out);
}
